// LMAttention_18803366821904
// MI455X (gfx1250) — matrix operand order repaired and checked by reading, not compiled, not run on hardware
//
#include <hip/hip_runtime.h>
#include <hip/hip_bf16.h>
#include <math.h>

typedef __attribute__((ext_vector_type(16))) _Float16 v16h;
typedef __attribute__((ext_vector_type(8)))  _Float16 v8h;
typedef __attribute__((ext_vector_type(4)))  _Float16 v4h;
typedef __attribute__((ext_vector_type(8)))  float    v8f;
typedef __attribute__((ext_vector_type(4)))  unsigned int v4u;
typedef __attribute__((ext_vector_type(8)))  int      v8i;
typedef __attribute__((ext_vector_type(4)))  int      v4i;

#define NEG_BIG (-1.0e30f)

__device__ __forceinline__ v8f zero_v8f() {
  v8f z;
#pragma unroll
  for (int i = 0; i < 8; ++i) z[i] = 0.0f;
  return z;
}

__device__ __forceinline__ v16h zero_v16h() {
  v16h z;
#pragma unroll
  for (int i = 0; i < 16; ++i) z[i] = (_Float16)0.0f;
  return z;
}

__device__ __forceinline__ v16h combine8(v8h lo, v8h hi) {
  v16h r;
#pragma unroll
  for (int i = 0; i < 8; ++i) { r[i] = lo[i]; r[i + 8] = hi[i]; }
  return r;
}

// 8 consecutive f32 -> 8 halves (p must be 16B aligned; works for global or LDS)
__device__ __forceinline__ v8h ld8f_to_h(const float* p) {
  float4 a = *(const float4*)p;
  float4 b = *(const float4*)(p + 4);
  v8h r;
  r[0] = (_Float16)a.x; r[1] = (_Float16)a.y; r[2] = (_Float16)a.z; r[3] = (_Float16)a.w;
  r[4] = (_Float16)b.x; r[5] = (_Float16)b.y; r[6] = (_Float16)b.z; r[7] = (_Float16)b.w;
  return r;
}

// 16 consecutive f32 -> v16h with r[j] = p[j]
__device__ __forceinline__ v16h ld16f_to_h(const float* p) {
  v16h r;
#pragma unroll
  for (int i = 0; i < 16; i += 4) {
    float4 x = *(const float4*)(p + i);
    r[i + 0] = (_Float16)x.x; r[i + 1] = (_Float16)x.y;
    r[i + 2] = (_Float16)x.z; r[i + 3] = (_Float16)x.w;
  }
  return r;
}

__device__ __forceinline__ float gelu_f(float x) {
  return 0.5f * x * (1.0f + erff(x * 0.70710678118654752f));
}

// ---------------------------------------------------------------------------
// prep: hidden[S,B,2H] -> qbf[M,2H], kvf[M,H], kvb[M,H]   (M = B*S2, [b][s])
// ---------------------------------------------------------------------------
__global__ __launch_bounds__(256) void prep_kernel(
    const float* __restrict__ hidden, float* __restrict__ qbf,
    float* __restrict__ kvf, float* __restrict__ kvb,
    int S2, int B, int H) {
  int idx = blockIdx.x * blockDim.x + threadIdx.x;
  int M = B * S2;
  if (idx >= M * H) return;
  int d = idx % H;
  int m = idx / H;
  int s = m % S2;
  int b = m / S2;
  int D2 = 2 * H;
  float fv = hidden[(size_t)s * (B * D2) + (size_t)b * D2 + d];
  float bv = hidden[(size_t)(s + 2) * (B * D2) + (size_t)b * D2 + H + d];
  qbf[(size_t)m * D2 + d] = fv;
  qbf[(size_t)m * D2 + H + d] = bv;
  kvf[(size_t)m * H + d] = fv;
  kvb[(size_t)m * H + d] = bv;
}

// ---------------------------------------------------------------------------
// GEMM: C[M,N] = act(A[M,K] @ W[K,N] + bias), f16 WMMA, f32 accumulate
// block = 128 threads (4 waves), tile 64x64, BK = 32.
// A tile (64x32 f32, row stride K) is staged into LDS by the Tensor Data
// Mover (tensor_load_to_lds) issued by wave 0; TDM OOB rows return zero,
// giving free zero-fill at the M boundary.
// ---------------------------------------------------------------------------
template <int ACT>
__global__ __launch_bounds__(128) void gemm_kernel(
    const float* __restrict__ A, const float* __restrict__ W,
    const float* __restrict__ bias, float* __restrict__ C,
    int M, int N, int K) {
  __shared__ __attribute__((aligned(16))) float    sAf[64 * 32];  // [row][k] f32 (TDM dest)
  __shared__ __attribute__((aligned(32))) _Float16 sW[64 * 32];   // [n][k] f16 (transposed)

  const int tid = threadIdx.x;
  const int lane = tid & 31;
  const int wid = tid >> 5;
  const int l15 = lane & 15;
  const int hiL = lane >> 4;
  const int kbase = hiL * 16;   // A-frag K offset (contiguous layout)
  const int kboff = hiL * 16;   // B-frag K offset (contiguous layout)
  const int m0 = blockIdx.y * 64;
  const int n0 = blockIdx.x * 64;

  int rows = M - m0;
  if (rows > 64) rows = 64;
  const unsigned ldsA = (unsigned)(size_t)(void*)sAf;  // LDS byte offset (addr[31:0])

  v8f acc[4];
#pragma unroll
  for (int nt = 0; nt < 4; ++nt) acc[nt] = zero_v8f();

  for (int k0 = 0; k0 < K; k0 += 32) {
    // ---- TDM: DMA the A tile (rows x 32 f32) global -> LDS, wave 0 only ----
    if (tid < 32) {
      unsigned long long gaddr =
          (unsigned long long)(size_t)(A + (size_t)m0 * K + k0);
      v4u g0;
      g0[0] = 1u;                                   // count=1, user descriptor
      g0[1] = ldsA;                                 // lds_addr
      g0[2] = (unsigned)(gaddr & 0xFFFFFFFFull);    // global_addr[31:0]
      g0[3] = (unsigned)((gaddr >> 32) & 0x1FFFFFFull) | (2u << 30); // [56:32]|type=2
      v8i g1;
      g1[0] = (int)(2u << 16);                      // data_size=4B (code 2)
      g1[1] = (int)(32u << 16);                     // tensor_dim0[15:0]=32
      g1[2] = (int)((unsigned)rows << 16);          // tensor_dim1[15:0]=rows
      g1[3] = (int)(32u << 16);                     // tile_dim0=32
      g1[4] = 64;                                   // tile_dim1=64, tile_dim2=0
      g1[5] = K;                                    // tensor_dim0_stride[31:0]=K
      g1[6] = 0;                                    // stride hi, dim1_stride lo
      g1[7] = 0;
      v4i gz;
      gz[0] = 0; gz[1] = 0; gz[2] = 0; gz[3] = 0;   // groups 2/3: dims beyond 2D unused
      v8i gz8;
#pragma unroll
      for (int i = 0; i < 8; ++i) gz8[i] = 0;
      __builtin_amdgcn_tensor_load_to_lds(g0, g1, gz, gz, gz8, 0);
      __builtin_amdgcn_s_wait_tensorcnt(0);
    }
    // ---- stage W tile 32x64 transposed -> sW[n][k] (f32 -> f16) ----
#pragma unroll
    for (int it = 0; it < 4; ++it) {
      int kk = (tid >> 4) + it * 8;
      int n4 = (tid & 15) * 4;
      float4 v = *(const float4*)(W + (size_t)(k0 + kk) * N + n0 + n4);
      sW[(n4 + 0) * 32 + kk] = (_Float16)v.x;
      sW[(n4 + 1) * 32 + kk] = (_Float16)v.y;
      sW[(n4 + 2) * 32 + kk] = (_Float16)v.z;
      sW[(n4 + 3) * 32 + kk] = (_Float16)v.w;
    }
    // prefetch next W K-tile while we compute on this one
    if (k0 + 32 < K) {
      __builtin_prefetch(W + (size_t)(k0 + 32 + (tid >> 2)) * N + n0, 0, 1);
    }
    __syncthreads();

    // A fragment from f32 LDS tile (contiguous A layout)
    int arow = wid * 16 + l15;
    const float* ap = &sAf[arow * 32];
    v16h afrag = combine8(ld8f_to_h(ap + kbase), ld8f_to_h(ap + 8 + kbase));
#pragma unroll
    for (int nt = 0; nt < 4; ++nt) {
      int bcol = nt * 16 + l15;
      v16h bfrag = *(const v16h*)(&sW[bcol * 32 + kboff]);
      acc[nt] = __builtin_amdgcn_wmma_f32_16x16x32_f16(
          false, afrag, false, bfrag, (short)0, acc[nt], false, false);
    }
    __syncthreads();
  }

  // epilogue: bias (+ GELU) and store
#pragma unroll
  for (int nt = 0; nt < 4; ++nt) {
    int gc = n0 + nt * 16 + l15;
    float bv = bias[gc];
#pragma unroll
    for (int r = 0; r < 8; ++r) {
      int gr = m0 + wid * 16 + hiL * 8 + r;
      if (gr < M) {
        float v = acc[nt][r] + bv;
        if (ACT) v = gelu_f(v);
        C[(size_t)gr * N + gc] = v;
      }
    }
  }
}

// ---------------------------------------------------------------------------
// Flash attention: one wave per (16-row q tile, head, batch).
// Q,K,V: [M, 512] f32 (M = B*S2, head h occupies cols h*64..h*64+63)
// CTX out: same layout. Scores scaled by 1/8 (dh=64).
// ---------------------------------------------------------------------------
__global__ __launch_bounds__(32) void attn_kernel(
    const float* __restrict__ Q, const float* __restrict__ Kt,
    const float* __restrict__ V, const int* __restrict__ lens,
    float* __restrict__ CTX, int S2, int backward) {
  __shared__ __attribute__((aligned(32))) _Float16 sP[16 * 32];

  const int lane = threadIdx.x;
  const int l15 = lane & 15;
  const int hiL = lane >> 4;
  const int kbase = hiL * 16;   // A-frag K offset
  const int kboff = hiL * 16;   // B-frag K offset
  const int qt = blockIdx.x;
  const int h = blockIdx.y;
  const int b = blockIdx.z;
  const int Lm = lens[b] - 2;
  const int H = 512;
  const size_t rowBase = (size_t)b * S2;
  const size_t hoff = (size_t)h * 64;

  // Q A-fragments (d 0..31 and 32..63), contiguous A layout
  int qrow = qt * 16 + l15;
  v16h qa0, qa1;
  if (qrow < S2) {
    const float* qp = Q + (rowBase + qrow) * H + hoff;
    qa0 = combine8(ld8f_to_h(qp + kbase), ld8f_to_h(qp + 8 + kbase));
    qa1 = combine8(ld8f_to_h(qp + 32 + kbase), ld8f_to_h(qp + 40 + kbase));
  } else {
    qa0 = zero_v16h();
    qa1 = zero_v16h();
  }

  v8f acc[4];
#pragma unroll
  for (int nt = 0; nt < 4; ++nt) acc[nt] = zero_v8f();
  float mrow[8], lrow[8];
#pragma unroll
  for (int r = 0; r < 8; ++r) { mrow[r] = NEG_BIG; lrow[r] = 0.0f; }

  const int kend = (S2 + 31) & ~31;
  for (int kt0 = 0; kt0 < kend; kt0 += 32) {
    // ---- scores for 32 kv positions: two 16x16 tiles ----
    v8f s0 = zero_v8f(), s1 = zero_v8f();
    {
      int kc = kt0 + l15;
      v16h kb0, kb1;
      if (kc < S2) {
        const float* kp = Kt + (rowBase + kc) * H + hoff;
        kb0 = ld16f_to_h(kp + kboff);        // d = kboff..kboff+15
        kb1 = ld16f_to_h(kp + 32 + kboff);   // d = 32+kboff..
      } else {
        kb0 = zero_v16h(); kb1 = zero_v16h();
      }
      s0 = __builtin_amdgcn_wmma_f32_16x16x32_f16(false, qa0, false, kb0, (short)0, s0, false, false);
      s0 = __builtin_amdgcn_wmma_f32_16x16x32_f16(false, qa1, false, kb1, (short)0, s0, false, false);
    }
    {
      int kc = kt0 + 16 + l15;
      v16h kb0, kb1;
      if (kc < S2) {
        const float* kp = Kt + (rowBase + kc) * H + hoff;
        kb0 = ld16f_to_h(kp + kboff);
        kb1 = ld16f_to_h(kp + 32 + kboff);
      } else {
        kb0 = zero_v16h(); kb1 = zero_v16h();
      }
      s1 = __builtin_amdgcn_wmma_f32_16x16x32_f16(false, qa0, false, kb0, (short)0, s1, false, false);
      s1 = __builtin_amdgcn_wmma_f32_16x16x32_f16(false, qa1, false, kb1, (short)0, s1, false, false);
    }

    // ---- masked online softmax (row stats per 16-lane half) ----
#pragma unroll
    for (int r = 0; r < 8; ++r) {
      int qi = qt * 16 + hiL * 8 + r;
      int ki0 = kt0 + l15;
      int ki1 = ki0 + 16;
      float sv0 = s0[r] * 0.125f;
      float sv1 = s1[r] * 0.125f;
      bool in0 = (ki0 < S2) && (qi < S2);
      bool in1 = (ki1 < S2) && (qi < S2);
      bool a0, a1;
      if (backward) {
        a0 = in0 && ((qi >= Lm) || ((ki0 >= qi) && (ki0 < Lm)));
        a1 = in1 && ((qi >= Lm) || ((ki1 >= qi) && (ki1 < Lm)));
      } else {
        a0 = in0 && ((qi >= Lm) || (ki0 <= qi));
        a1 = in1 && ((qi >= Lm) || (ki1 <= qi));
      }
      sv0 = a0 ? sv0 : NEG_BIG;
      sv1 = a1 ? sv1 : NEG_BIG;
      float mx = fmaxf(sv0, sv1);
#pragma unroll
      for (int o = 1; o < 16; o <<= 1) mx = fmaxf(mx, __shfl_xor(mx, o, 32));
      float mn = fmaxf(mrow[r], mx);
      float e0 = a0 ? __expf(sv0 - mn) : 0.0f;
      float e1 = a1 ? __expf(sv1 - mn) : 0.0f;
      float rs = e0 + e1;
#pragma unroll
      for (int o = 1; o < 16; o <<= 1) rs += __shfl_xor(rs, o, 32);
      float scl = __expf(mrow[r] - mn);
      lrow[r] = lrow[r] * scl + rs;
      mrow[r] = mn;
#pragma unroll
      for (int nt = 0; nt < 4; ++nt) acc[nt][r] *= scl;
      sP[(r + hiL * 8) * 32 + l15] = (_Float16)e0;
      sP[(r + hiL * 8) * 32 + 16 + l15] = (_Float16)e1;
    }
    asm volatile("s_wait_dscnt 0" ::: "memory");   // LDS C->A relayout visible

    // ---- ctx += P(16x32) @ V(32x64) ----
    {
      const _Float16* pr = &sP[l15 * 32];
      v8h plo = *(const v8h*)(pr + kbase);
      v8h phi = *(const v8h*)(pr + 8 + kbase);
      v16h pa = combine8(plo, phi);
#pragma unroll
      for (int nt = 0; nt < 4; ++nt) {
        v16h vb;
#pragma unroll
        for (int j = 0; j < 16; ++j) {
          int kk = kt0 + kboff + j;
          if (kk > S2 - 1) kk = S2 - 1;   // P is zero there; clamp keeps reads in-bounds
          vb[j] = (_Float16)V[(rowBase + kk) * H + hoff + nt * 16 + l15];
        }
        acc[nt] = __builtin_amdgcn_wmma_f32_16x16x32_f16(
            false, pa, false, vb, (short)0, acc[nt], false, false);
      }
    }
  }

  // ---- normalize and store ----
#pragma unroll
  for (int nt = 0; nt < 4; ++nt) {
#pragma unroll
    for (int r = 0; r < 8; ++r) {
      int row = qt * 16 + hiL * 8 + r;
      if (row < S2) {
        CTX[(rowBase + row) * H + hoff + nt * 16 + l15] = acc[nt][r] / lrow[r];
      }
    }
  }
}

// ---------------------------------------------------------------------------
// LayerNorm over last dim D (256 threads per row). Optional (S2,B) transpose
// of the output for the final result.
// ---------------------------------------------------------------------------
__global__ __launch_bounds__(256) void ln_kernel(
    const float* __restrict__ X, const float* __restrict__ gam,
    const float* __restrict__ bet, float* __restrict__ Y,
    int D, int S2, int Bn, int transposeOut) {
  __shared__ float red[256];
  int m = blockIdx.x;
  int tid = threadIdx.x;
  const float* x = X + (size_t)m * D;
  int nc = D >> 8;  // 2 or 4
  float xv[4];
  float s = 0.0f;
  for (int c = 0; c < nc; ++c) { xv[c] = x[tid + c * 256]; s += xv[c]; }
  red[tid] = s;
  __syncthreads();
  for (int off = 128; off > 0; off >>= 1) {
    if (tid < off) red[tid] += red[tid + off];
    __syncthreads();
  }
  float mean = red[0] / (float)D;
  __syncthreads();
  float s2 = 0.0f;
  for (int c = 0; c < nc; ++c) { float dlt = xv[c] - mean; s2 += dlt * dlt; }
  red[tid] = s2;
  __syncthreads();
  for (int off = 128; off > 0; off >>= 1) {
    if (tid < off) red[tid] += red[tid + off];
    __syncthreads();
  }
  float inv = rsqrtf(red[0] / (float)D + 1e-12f);
  for (int c = 0; c < nc; ++c) {
    int d = tid + c * 256;
    float yv = (xv[c] - mean) * inv * gam[d] + bet[d];
    if (!transposeOut) {
      Y[(size_t)m * D + d] = yv;
    } else {
      int sI = m % S2, bI = m / S2;
      Y[((size_t)sI * Bn + bI) * D + d] = yv;
    }
  }
}

// ---------------------------------------------------------------------------
// attn_out = LN(qbf + concat(f_attn, b_attn))   (D = 1024, H = 512)
// ---------------------------------------------------------------------------
__global__ __launch_bounds__(256) void resid_ln_kernel(
    const float* __restrict__ qbf, const float* __restrict__ fa,
    const float* __restrict__ ba, const float* __restrict__ gam,
    const float* __restrict__ bet, float* __restrict__ Y, int H) {
  __shared__ float red[256];
  const int D = 2 * H;
  int m = blockIdx.x;
  int tid = threadIdx.x;
  float xv[4];
  float s = 0.0f;
#pragma unroll
  for (int c = 0; c < 4; ++c) {
    int d = tid + c * 256;
    float add = (d < H) ? fa[(size_t)m * H + d] : ba[(size_t)m * H + d - H];
    xv[c] = qbf[(size_t)m * D + d] + add;
    s += xv[c];
  }
  red[tid] = s;
  __syncthreads();
  for (int off = 128; off > 0; off >>= 1) {
    if (tid < off) red[tid] += red[tid + off];
    __syncthreads();
  }
  float mean = red[0] / (float)D;
  __syncthreads();
  float s2 = 0.0f;
#pragma unroll
  for (int c = 0; c < 4; ++c) { float dlt = xv[c] - mean; s2 += dlt * dlt; }
  red[tid] = s2;
  __syncthreads();
  for (int off = 128; off > 0; off >>= 1) {
    if (tid < off) red[tid] += red[tid + off];
    __syncthreads();
  }
  float inv = rsqrtf(red[0] / (float)D + 1e-12f);
#pragma unroll
  for (int c = 0; c < 4; ++c) {
    int d = tid + c * 256;
    Y[(size_t)m * D + d] = (xv[c] - mean) * inv * gam[d] + bet[d];
  }
}

// ---------------------------------------------------------------------------
extern "C" void kernel_launch(void* const* d_in, const int* in_sizes, int n_in,
                              void* d_out, int out_size, void* d_ws, size_t ws_size,
                              hipStream_t stream) {
  (void)in_sizes; (void)n_in; (void)out_size; (void)ws_size;
  const int S2 = 1022, B = 8, H = 512;
  const int M = B * S2;              // 8176
  const int D2 = 2 * H;              // 1024

  const float* hidden = (const float*)d_in[0];
  const int* lens = (const int*)d_in[1];
  auto F = [&](int i) { return (const float*)d_in[i]; };
  // f block: 2..15, b block: 16..29
  const int FO = 2, BO = 16;

  float* ws = (float*)d_ws;
  const size_t R1024 = (size_t)8192 * 1024;
  const size_t R512 = (size_t)8192 * 512;
  float* qbf = ws;                 // [M,1024]  (later reused for out-FFN t1)
  float* h1 = ws + R1024;          // [M,1024]  (FFN hidden, attn_out, t2)
  float* kvf = ws + 2 * R1024;     // [M,512]
  float* kvb = kvf + R512;         // [M,512]
  float* tmp = kvb + R512;         // [M,512]   (fq / bq)
  float* qp = tmp + R512;          // [M,512]
  float* kp = qp + R512;           // [M,512]
  float* vp = kp + R512;           // [M,512]
  float* ctx = vp + R512;          // [M,512]
  float* fat = ctx + R512;         // [M,512]   f_attn
  float* bat = fat + R512;         // [M,512]   b_attn
  float* outp = (float*)d_out;     // [S2,B,1024]

  int total = M * H;
  prep_kernel<<<(total + 255) / 256, 256, 0, stream>>>(hidden, qbf, kvf, kvb, S2, B, H);

  dim3 blk(128);
  dim3 gN1024(1024 / 64, (M + 63) / 64);
  dim3 gN512(512 / 64, (M + 63) / 64);
  dim3 gAttn((S2 + 15) / 16, 8, B);

  // ---- forward direction ----
  gemm_kernel<1><<<gN1024, blk, 0, stream>>>(qbf, F(FO + 0), F(FO + 1), h1, M, 1024, 1024);
  gemm_kernel<0><<<gN512, blk, 0, stream>>>(h1, F(FO + 2), F(FO + 3), tmp, M, 512, 1024);
  ln_kernel<<<M, 256, 0, stream>>>(tmp, F(FO + 4), F(FO + 5), tmp, 512, S2, B, 0);
  gemm_kernel<0><<<gN512, blk, 0, stream>>>(tmp, F(FO + 6), F(FO + 7), qp, M, 512, 512);
  gemm_kernel<0><<<gN512, blk, 0, stream>>>(kvf, F(FO + 8), F(FO + 9), kp, M, 512, 512);
  gemm_kernel<0><<<gN512, blk, 0, stream>>>(kvf, F(FO + 10), F(FO + 11), vp, M, 512, 512);
  attn_kernel<<<gAttn, 32, 0, stream>>>(qp, kp, vp, lens, ctx, S2, 0);
  gemm_kernel<0><<<gN512, blk, 0, stream>>>(ctx, F(FO + 12), F(FO + 13), fat, M, 512, 512);

  // ---- backward direction ----
  gemm_kernel<1><<<gN1024, blk, 0, stream>>>(qbf, F(BO + 0), F(BO + 1), h1, M, 1024, 1024);
  gemm_kernel<0><<<gN512, blk, 0, stream>>>(h1, F(BO + 2), F(BO + 3), tmp, M, 512, 1024);
  ln_kernel<<<M, 256, 0, stream>>>(tmp, F(BO + 4), F(BO + 5), tmp, 512, S2, B, 0);
  gemm_kernel<0><<<gN512, blk, 0, stream>>>(tmp, F(BO + 6), F(BO + 7), qp, M, 512, 512);
  gemm_kernel<0><<<gN512, blk, 0, stream>>>(kvb, F(BO + 8), F(BO + 9), kp, M, 512, 512);
  gemm_kernel<0><<<gN512, blk, 0, stream>>>(kvb, F(BO + 10), F(BO + 11), vp, M, 512, 512);
  attn_kernel<<<gAttn, 32, 0, stream>>>(qp, kp, vp, lens, ctx, S2, 1);
  gemm_kernel<0><<<gN512, blk, 0, stream>>>(ctx, F(BO + 12), F(BO + 13), bat, M, 512, 512);

  // ---- residual + concat + LN -> attn_out (h1) ----
  resid_ln_kernel<<<M, 256, 0, stream>>>(qbf, fat, bat, F(30), F(31), h1, H);

  // ---- output FFN + final LN (transposed to [S2,B,1024]) ----
  gemm_kernel<1><<<gN1024, blk, 0, stream>>>(h1, F(32), F(33), qbf, M, 1024, 1024);
  gemm_kernel<0><<<gN1024, blk, 0, stream>>>(qbf, F(34), F(35), h1, M, 1024, 1024);
  ln_kernel<<<M, 256, 0, stream>>>(h1, F(36), F(37), outp, D2, S2, B, 1);
}
